// NucleusSamplingGenerator_9345848836436
// MI455X (gfx1250) — matrix operand order repaired and checked by reading, not compiled, not run on hardware
//
#include <hip/hip_runtime.h>
#include <hip/hip_bf16.h>
#include <stdint.h>

// ---------------------------------------------------------------------------
// CDNA5 / gfx1250, wave32.  Pipeline:
//   K1 pack_x_bf16 : x fp32 -> bf16, pre-swizzled into WMMA A-fragment order
//   K2 gemm_wmma   : logits = x @ W + b, W converted fp32->bf16 on the fly
//                    (single HBM pass, NT hint), v_wmma_f32_16x16x32_bf16,
//                    double-buffered LDS B-tiles, A-loads issued before W-loads
//                    so WMMA can proceed at loadcnt<=8 while W streams in.
//   K3 topp_sample : per-row softmax + nucleus threshold (binary search in
//                    320KB LDS, no sort) + Threefry2x32 Gumbel argmax (key 42)
// ---------------------------------------------------------------------------

typedef __attribute__((ext_vector_type(16))) __bf16 v16bf;
typedef __attribute__((ext_vector_type(8)))  float  v8f;

__device__ __forceinline__ unsigned f2bf_bits(float f) {
  unsigned u = __float_as_uint(f);
  return (u + 0x7FFFu + ((u >> 16) & 1u)) >> 16;     // round-to-nearest-even
}
__device__ __forceinline__ unsigned short f2bf_rne(float f) {
  return (unsigned short)f2bf_bits(f);
}
__device__ __forceinline__ unsigned pk_bf2(float a, float b) {
  return f2bf_bits(a) | (f2bf_bits(b) << 16);
}

// ---------------------------------------------------------------------------
// K1: pack x[64][D] fp32 into bf16 A-fragments.
// A 16x32 bf16 fragment (ISA 7.12.2): lane L holds row M=L%16,
// element i -> k = (i&7) + (L<16?0:8) + (i>=8?16:0)   (within a K32 chunk)
// Flat fragment index f = (((kc*4 + mr)*32 + L)*16 + i)
// ---------------------------------------------------------------------------
__global__ void pack_x_bf16(const float* __restrict__ x,
                            unsigned short* __restrict__ xfrag,
                            int D, int total) {
  int f = blockIdx.x * blockDim.x + threadIdx.x;
  if (f >= total) return;
  int i  = f & 15;
  int L  = (f >> 4) & 31;
  int mr = (f >> 9) & 3;
  int kc = f >> 11;
  int m     = mr * 16 + (L & 15);
  int kbase = (L < 16) ? 0 : 8;
  int k     = kc * 32 + (i & 7) + kbase + ((i >= 8) ? 16 : 0);
  xfrag[f] = f2bf_rne(x[(long)m * D + k]);
}

// ---------------------------------------------------------------------------
// K2: GEMM.  Block = 256 threads (8 waves): M=64 (all rows) x N=64 cols,
// K chunks of 32, double-buffered 4KB LDS B-tile.
// Staging map: thread t -> column c = t&63 (clamped to V-1: pad columns are
// staged but never stored), k = (t>>6)*8 .. +7 -> ONE ds_store_b128.
// B 32x16 fragment: lane = n%16 + ((k>>3)&1?16:0), element i = (k>>4)*8 + (k&7).
// Wave w: M-tile mr=w&3, N-groups g0=(w>>2)*2, g0+1 -> 2 WMMAs/chunk.
// CV/CD: compile-time V/D (0 = dynamic).  Compile-time V folds the 8 strided
// W loads into immediate offsets (r*V*4 < 2^23 fits signed 24-bit IOFFSET).
// ---------------------------------------------------------------------------
template <int CV, int CD>
__global__ __launch_bounds__(256) void gemm_wmma(
    const unsigned short* __restrict__ xfrag, const float* __restrict__ W,
    const float* __restrict__ bias, float* __restrict__ logits,
    int Vdyn, int Ddyn, int Vpad) {
  const int V = CV ? CV : Vdyn;
  const int D = CD ? CD : Ddyn;
  __shared__ __align__(32) unsigned short Btile[2][4 * 32 * 16];  // 2 x 4 KB
  const int tid  = threadIdx.x;
  const int lane = tid & 31;
  const int wv   = tid >> 5;
  const int mr   = wv & 3;
  const int g0   = (wv >> 2) * 2;
  const long n0  = (long)blockIdx.x * 64;
  const int  Kc  = D >> 5;

  // ---- staging-role constants ----
  const int  c   = tid & 63;          // column within 64-wide tile
  const int  kq  = tid >> 6;          // 0..3 -> k = kq*8 .. kq*8+7
  long ncl = n0 + c;                  // clamp: pad cols stage valid-but-unused data
  if (ncl > (long)(V - 1)) ncl = V - 1;
  const float* wp = W + (size_t)(kq * 8) * (size_t)V + ncl;
  const size_t kcStride = (size_t)32 * (size_t)V;     // floats per K-chunk
  const int fragOff = ((c >> 4) * 32 + (c & 15) + ((kq & 1) ? 16 : 0)) * 16
                      + (kq >> 1) * 8;
  uint4* const dstA = reinterpret_cast<uint4*>(&Btile[0][fragOff]);
  uint4* const dstB = reinterpret_cast<uint4*>(&Btile[1][fragOff]);

  // ---- compute-role constants ----
  const unsigned short* ap = xfrag + ((size_t)mr * 32 + lane) * 16;
  const int bOff0 = (g0 * 32 + lane) * 16;
  const int bOff1 = ((g0 + 1) * 32 + lane) * 16;

  v8f acc0 = {};
  v8f acc1 = {};
  float f[8];

  auto load8 = [&](int kc) {          // issue 8 coalesced NT loads (in flight)
    const float* p = wp + (size_t)kc * kcStride;
#pragma unroll
    for (int r = 0; r < 8; ++r)
      f[r] = __builtin_nontemporal_load(p + (size_t)r * (size_t)V);
  };
  auto store8 = [&](uint4* dst) {     // pack + one ds_store_b128 (late)
    uint4 u;
    u.x = pk_bf2(f[0], f[1]);
    u.y = pk_bf2(f[2], f[3]);
    u.z = pk_bf2(f[4], f[5]);
    u.w = pk_bf2(f[6], f[7]);
    *dst = u;
  };

  // prologue: fully stage chunk 0 into buffer A
  load8(0);
  store8(dstA);
  int pb = 0;

  // steady state: A loads first, then next-chunk W loads, WMMA (W still in
  // flight), then pack+store staged tile into the other buffer.
  for (int kc = 0; kc < Kc - 1; ++kc) {
    __syncthreads();
    v16bf a = *reinterpret_cast<const v16bf*>(ap);   // A loads issued first
    ap += 4 * 32 * 16;
    load8(kc + 1);                                   // W loads issued second
    const unsigned short* bb = &Btile[pb][0];
    v16bf b0 = *reinterpret_cast<const v16bf*>(bb + bOff0);
    v16bf b1 = *reinterpret_cast<const v16bf*>(bb + bOff1);
    acc0 = __builtin_amdgcn_wmma_f32_16x16x32_bf16(false, a, false, b0,
                                                   (short)0, acc0, false, false);
    acc1 = __builtin_amdgcn_wmma_f32_16x16x32_bf16(false, a, false, b1,
                                                   (short)0, acc1, false, false);
    store8(pb ? dstA : dstB);
    pb ^= 1;
  }
  // epilogue chunk (no staging)
  __syncthreads();
  {
    v16bf a = *reinterpret_cast<const v16bf*>(ap);
    const unsigned short* bb = &Btile[pb][0];
    v16bf b0 = *reinterpret_cast<const v16bf*>(bb + bOff0);
    v16bf b1 = *reinterpret_cast<const v16bf*>(bb + bOff1);
    acc0 = __builtin_amdgcn_wmma_f32_16x16x32_bf16(false, a, false, b0,
                                                   (short)0, acc0, false, false);
    acc1 = __builtin_amdgcn_wmma_f32_16x16x32_bf16(false, a, false, b1,
                                                   (short)0, acc1, false, false);
  }

  // C/D layout: VGPR j -> row = j + (lane>=16 ? 8 : 0), col = lane%16
  int mlo = mr * 16 + ((lane >> 4) * 8);
  int nc  = lane & 15;
#pragma unroll
  for (int j = 0; j < 8; ++j) {
    long n1 = n0 + g0 * 16 + nc;
    long n2 = n0 + (g0 + 1) * 16 + nc;
    if (n1 < V) logits[(long)(mlo + j) * Vpad + n1] = acc0[j] + bias[n1];
    if (n2 < V) logits[(long)(mlo + j) * Vpad + n2] = acc1[j] + bias[n2];
  }
}

// ---------------------------------------------------------------------------
// K3 helpers
// ---------------------------------------------------------------------------
__device__ __forceinline__ float block_max(float v, volatile float* red) {
  for (int o = 16; o; o >>= 1) v = fmaxf(v, __shfl_xor(v, o, 32));
  __syncthreads();
  if ((threadIdx.x & 31) == 0) red[threadIdx.x >> 5] = v;
  __syncthreads();
  float r = -INFINITY;
  int nw = blockDim.x >> 5;
  for (int i = 0; i < nw; ++i) r = fmaxf(r, red[i]);
  __syncthreads();
  return r;
}

__device__ __forceinline__ float block_sum(float v, volatile float* red) {
  for (int o = 16; o; o >>= 1) v += __shfl_xor(v, o, 32);
  __syncthreads();
  if ((threadIdx.x & 31) == 0) red[threadIdx.x >> 5] = v;
  __syncthreads();
  float r = 0.0f;
  int nw = blockDim.x >> 5;
  for (int i = 0; i < nw; ++i) r += red[i];
  __syncthreads();
  return r;
}

// Threefry-2x32 with JAX key schedule, key = PRNGKey(42) = (0, 42).
// Flat counter i in [0, n): block c = i%half with (x0,x1)=(c, c+half);
// output = out0 if i<half else out1.  Then JAX uniform(tiny,1) -> Gumbel.
#define TF_RND(r) { x0 += x1; x1 = (x1 << r) | (x1 >> (32 - r)); x1 ^= x0; }
__device__ __forceinline__ float gumbel42(unsigned idx, unsigned n_total) {
  const unsigned half = n_total >> 1;
  unsigned c  = (idx < half) ? idx : idx - half;
  unsigned x0 = c, x1 = c + half;
  const unsigned k0 = 0u, k1 = 42u;
  const unsigned k2 = 0x1BD11BDAu ^ k0 ^ k1;
  x0 += k0; x1 += k1;
  TF_RND(13) TF_RND(15) TF_RND(26) TF_RND(6)   x0 += k1; x1 += k2 + 1u;
  TF_RND(17) TF_RND(29) TF_RND(16) TF_RND(24)  x0 += k2; x1 += k0 + 2u;
  TF_RND(13) TF_RND(15) TF_RND(26) TF_RND(6)   x0 += k0; x1 += k1 + 3u;
  TF_RND(17) TF_RND(29) TF_RND(16) TF_RND(24)  x0 += k1; x1 += k2 + 4u;
  TF_RND(13) TF_RND(15) TF_RND(26) TF_RND(6)   x0 += k2; x1 += k0 + 5u;
  unsigned bits = (idx < half) ? x0 : x1;
  float f = __uint_as_float((bits >> 9) | 0x3F800000u) - 1.0f;  // [0,1)
  const float tiny = 1.1754944e-38f;
  float u = fmaxf(tiny, f * (1.0f - tiny) + tiny);
  return -logf(-logf(u));
}

// ---------------------------------------------------------------------------
// K3: one block per row.  e[] = exp(l - max) lives entirely in LDS (201 KB,
// CDNA5 allows 320 KB/WG).  Token kept iff sum{e' > e} <= 0.9*Z, found by
// binary-searching the threshold (equivalent to the sort+cumsum mask, modulo
// exact-float ties).  Then Gumbel-argmax over kept tokens.
// ---------------------------------------------------------------------------
__global__ __launch_bounds__(1024) void topp_sample(
    const float* __restrict__ logits, float* __restrict__ out,
    int V, int Vpad, unsigned n_total, float cum_p) {
  extern __shared__ float e[];   // V floats
  __shared__ float redf[32];
  __shared__ float reds[32];
  __shared__ int   redi[32];
  const int row = blockIdx.x;
  const int tid = threadIdx.x;
  const float* lr = logits + (long)row * Vpad;

  // 1) max logit
  float m = -INFINITY;
  for (int v = tid; v < V; v += blockDim.x) m = fmaxf(m, lr[v]);
  m = block_max(m, redf);

  // 2) exponentials + partition function
  float zs = 0.0f;
  for (int v = tid; v < V; v += blockDim.x) {
    float t = expf(lr[v] - m);
    e[v] = t;
    zs += t;
  }
  float Z = block_sum(zs, redf);
  float target = cum_p * Z;

  // 3) binary search nucleus threshold theta in (0, 1]:
  //    invariant G(lo)=sum{e>lo} > target >= G(hi)
  float lo = 0.0f, hi = 1.0f;
  for (int it = 0; it < 40; ++it) {
    float mid = 0.5f * (lo + hi);
    float g = 0.0f;
    for (int v = tid; v < V; v += blockDim.x) {
      float t = e[v];
      if (t > mid) g += t;
    }
    g = block_sum(g, redf);
    if (g <= target) hi = mid; else lo = mid;
  }
  // kept(v) := e[v] > lo

  // 4) kept mass; p'_v = e_v / (S + 1e-6*Z)  (== reference renorm in p-space)
  float ks = 0.0f;
  for (int v = tid; v < V; v += blockDim.x) {
    float t = e[v];
    if (t > lo) ks += t;
  }
  float S = block_sum(ks, redf);
  float inv = 1.0f / (S + 1e-6f * Z);

  // 5) categorical via Gumbel argmax over kept tokens
  float bs = -INFINITY;
  int   bi = 0x7FFFFFFF;
  for (int v = tid; v < V; v += blockDim.x) {
    float t = e[v];
    if (t > lo) {
      float sc = logf(t * inv) + gumbel42((unsigned)row * (unsigned)V + (unsigned)v, n_total);
      if (sc > bs || (sc == bs && v < bi)) { bs = sc; bi = v; }
    }
  }
  for (int o = 16; o; o >>= 1) {
    float s2 = __shfl_xor(bs, o, 32);
    int   i2 = __shfl_xor(bi, o, 32);
    if (s2 > bs || (s2 == bs && i2 < bi)) { bs = s2; bi = i2; }
  }
  __syncthreads();
  if ((tid & 31) == 0) { reds[tid >> 5] = bs; redi[tid >> 5] = bi; }
  __syncthreads();
  if (tid == 0) {
    float b = reds[0]; int ii = redi[0];
    for (int i = 1; i < (int)(blockDim.x >> 5); ++i)
      if (reds[i] > b || (reds[i] == b && redi[i] < ii)) { b = reds[i]; ii = redi[i]; }
    out[row] = (float)ii;
  }
}

// ---------------------------------------------------------------------------
extern "C" void kernel_launch(void* const* d_in, const int* in_sizes, int n_in,
                              void* d_out, int out_size, void* d_ws, size_t ws_size,
                              hipStream_t stream) {
  const float* x = (const float*)d_in[0];
  const float* W = (const float*)d_in[1];
  const float* b = (const float*)d_in[2];

  const int V = in_sizes[2];            // 50257
  const int D = in_sizes[1] / V;        // 2048
  const int B = in_sizes[0] / D;        // 64
  const int Vpad = ((V + 63) / 64) * 64;

  // workspace layout: [ x bf16 fragments | logits fp32 (B x Vpad) ]  ~13.1 MB
  unsigned short* xfrag = (unsigned short*)d_ws;
  float* logits = (float*)((char*)d_ws + (size_t)B * D * sizeof(unsigned short));

  const int packN = B * D;
  hipLaunchKernelGGL(pack_x_bf16, dim3((packN + 255) / 256), dim3(256), 0, stream,
                     x, xfrag, D, packN);

  if (V == 50257 && D == 2048) {
    hipLaunchKernelGGL((gemm_wmma<50257, 2048>), dim3(Vpad / 64), dim3(256), 0,
                       stream, xfrag, W, b, logits, V, D, Vpad);
  } else {
    hipLaunchKernelGGL((gemm_wmma<0, 0>), dim3(Vpad / 64), dim3(256), 0,
                       stream, xfrag, W, b, logits, V, D, Vpad);
  }

  const unsigned n_total = (unsigned)B * (unsigned)V;
  const size_t shmem = (size_t)V * sizeof(float);   // 201,028 B < 320 KB LDS
  hipLaunchKernelGGL(topp_sample, dim3(B), dim3(1024), shmem, stream,
                     logits, (float*)d_out, V, Vpad, n_total, 0.9f);
}
